// SimpleBiGRU_53747220742485
// MI455X (gfx1250) — matrix operand order repaired and checked by reading, not compiled, not run on hardware
//
#include <hip/hip_runtime.h>
#include <hip/hip_bf16.h>
#include <math.h>

// ---------------------------------------------------------------------------
// SimpleBiGRU on MI455X (gfx1250, wave32, WMMA).
//   B=64, T=512, D=512, H=512.  Output: [B,T,2H] f32.
//
// Phase 1: convert x / Wi / Wh to bf16 (Wi, Wh transposed to [N][K]).
// Phase 2: x_proj = x@Wi + bi  via v_wmma_f32_16x16x32_bf16.  Each wave
//          keeps its Wi column-tile (16 fragments, 128 VGPRs) register-
//          resident and streams 16 M-tiles of A through a depth-4 pipeline.
// Phase 3: persistent scan kernel, 96 blocks (48/dir), device barriers.
//          Wh fragments register-resident across all 512 timesteps; the h
//          (A) fragments stream through a depth-4 register pipeline.
// ---------------------------------------------------------------------------

#define BATCH 64
#define TLEN  512
#define DDIM  512
#define HDIM  512
#define N3H   1536            // 3*H
#define BT    (BATCH * TLEN)  // 32768
#define NTILES_N 96           // N3H / 16
#define MT_PER_WAVE 16        // M-tiles per wave in x_proj
#define SCAN_BLOCKS_PER_DIR 48
#define SCAN_THREADS 256

typedef __attribute__((ext_vector_type(16))) __bf16   v16bf;
typedef __attribute__((ext_vector_type(8)))  float    v8f;
typedef __attribute__((ext_vector_type(4)))  unsigned v4u;

union FragBF {
    v16bf v;
    v4u   q[2];
};

__device__ inline unsigned short f32_to_bf16_rne(float f) {
    unsigned u = __float_as_uint(f);
    unsigned r = u + 0x7FFFu + ((u >> 16) & 1u);   // round-to-nearest-even
    return (unsigned short)(r >> 16);
}

// A fragment: 16x32 bf16 tile, row-major source, leading dim ld (elements).
// lane l: row M = l&15 ; kb = (l>=16)?8:0 ; elems 0..7 = K kb..kb+7,
// elems 8..15 = K kb+16..kb+23   (ISA 7.12.2, 16-bit A 16x32).
__device__ inline v16bf load_a16x32(const unsigned short* __restrict__ base,
                                    int ld, int lane) {
    const int m  = lane & 15;
    const int kb = (lane >> 4) << 3;       // 0 or 8
    const unsigned short* p = base + (size_t)m * ld + kb;
    FragBF f;
    f.q[0] = *(const v4u*)(p);
    f.q[1] = *(const v4u*)(p + 16);
    return f.v;
}

// B fragment: 32x16 bf16 tile taken from W^T stored [N][K] row-major, ld=K.
// lane l: col N = l&15 ; kb = (l>=16)?8:0 ; elems 0..7 = K kb..kb+7,
// elems 8..15 = K kb+16..kb+23.
__device__ inline v16bf load_b32x16(const unsigned short* __restrict__ wT,
                                    int ld, int lane) {
    const int n  = lane & 15;
    const int kb = (lane >> 4) << 3;       // 0 or 8
    const unsigned short* p = wT + (size_t)n * ld + kb;
    FragBF f;
    f.q[0] = *(const v4u*)(p);
    f.q[1] = *(const v4u*)(p + 16);
    return f.v;
}

// ------------------------- conversion kernels ------------------------------

__global__ void k_init_bar(unsigned* bar) {
    if (threadIdx.x < 8) bar[threadIdx.x] = 0u;
}

__global__ void k_cvt_bf16(const float* __restrict__ src,
                           unsigned short* __restrict__ dst, int n) {
    for (int i = blockIdx.x * blockDim.x + threadIdx.x; i < n;
         i += gridDim.x * blockDim.x)
        dst[i] = f32_to_bf16_rne(src[i]);
}

// W [K][N] f32 (row-major)  ->  WT [N][K] bf16 (row-major)
__global__ void k_cvt_transpose(const float* __restrict__ W,
                                unsigned short* __restrict__ WT,
                                int K, int N) {
    int total = K * N;
    for (int i = blockIdx.x * blockDim.x + threadIdx.x; i < total;
         i += gridDim.x * blockDim.x) {
        int k = i / N;
        int n = i - k * N;
        WT[(size_t)n * K + k] = f32_to_bf16_rne(W[i]);
    }
}

// ------------------------- x_proj GEMM -------------------------------------
// C[32768 x 1536] = Abf[32768 x 512] * WiT^T + bi, per direction.
// grid = (1536, 1, 2), block = 256 (8 waves).
// Wave w: N-tile = w%96 (B fragments register-resident for the whole wave),
// M-chunk = w/96 -> 16 M-tiles of 16 rows, A streamed via depth-4 pipeline.
__global__ void __launch_bounds__(256, 1)
k_xproj(const unsigned short* __restrict__ xbf,
        const unsigned short* __restrict__ WiT_f,
        const unsigned short* __restrict__ WiT_b,
        const float* __restrict__ bi_f,
        const float* __restrict__ bi_b,
        float* __restrict__ xp_f,
        float* __restrict__ xp_b) {
    const int dir = blockIdx.z;
    const unsigned short* WT = dir ? WiT_b : WiT_f;
    const float* bi = dir ? bi_b : bi_f;
    float* xp = dir ? xp_b : xp_f;

    const int lane = threadIdx.x & 31;
    const int wave = threadIdx.x >> 5;
    const int w = blockIdx.x * 8 + wave;           // 0..12287
    const int ntile  = w % NTILES_N;               // 0..95
    const int mchunk = w / NTILES_N;               // 0..127
    const int n0 = ntile * 16;

    // hoist Wi column tile: 16 B fragments, register-resident
    const unsigned short* Bbase = WT + (size_t)n0 * DDIM;
    v16bf breg[16];
#pragma unroll
    for (int q = 0; q < 16; ++q)
        breg[q] = load_b32x16(Bbase + q * 32, DDIM, lane);

    const int col = n0 + (lane & 15);
    const float bv = bi[col];

    for (int mt = 0; mt < MT_PER_WAVE; ++mt) {
        const int m0 = mchunk * (MT_PER_WAVE * 16) + mt * 16;
        const unsigned short* Abase = xbf + (size_t)m0 * DDIM;

        v8f acc = {};
        v16bf areg[4];
#pragma unroll
        for (int p = 0; p < 4; ++p)
            areg[p] = load_a16x32(Abase + p * 32, DDIM, lane);
#pragma unroll
        for (int q = 0; q < 16; ++q) {
            v16bf cur = areg[q & 3];
            if (q + 4 < 16)        // refill this slot 4 WMMAs ahead
                areg[q & 3] = load_a16x32(Abase + (q + 4) * 32, DDIM, lane);
            acc = __builtin_amdgcn_wmma_f32_16x16x32_bf16(false, cur, false,
                                                          breg[q], (short)0,
                                                          acc, false, false);
        }

        const int rb = m0 + ((lane >> 4) << 3);
#pragma unroll
        for (int v = 0; v < 8; ++v)
            xp[(size_t)(rb + v) * N3H + col] = acc[v] + bv;
    }
}

// ------------------------- device-wide barrier -----------------------------

__device__ inline void grid_barrier(unsigned* cnt, unsigned* gen, unsigned nb) {
    __threadfence();          // make this wave's global writes device-visible
    __syncthreads();
    if (threadIdx.x == 0) {
        unsigned g = __atomic_load_n(gen, __ATOMIC_RELAXED);
        if (atomicAdd(cnt, 1u) == nb - 1u) {
            __atomic_store_n(cnt, 0u, __ATOMIC_RELAXED);
            __threadfence();
            atomicAdd(gen, 1u);
        } else {
            while (__atomic_load_n(gen, __ATOMIC_RELAXED) == g)
                __builtin_amdgcn_s_sleep(2);
        }
    }
    __syncthreads();
}

// ------------------------- persistent scan ---------------------------------
// grid = 96 blocks x 256 threads.  Blocks [0..48) = forward, [48..96) = bwd.
// Per step: 384 waves/dir cover the 4x96 16x16 tiles of hp = h_bf16 @ WhT.
// The 16 B (Wh) fragments per wave (128 VGPRs) stay register-resident for
// the entire 512-step loop; A (h) fragments stream through a depth-4
// register pipeline (8 b128 loads in flight under the WMMA chain).
__global__ void __launch_bounds__(256, 1)
k_scan(const float* __restrict__ xp_f,
       const float* __restrict__ xp_b,
       const unsigned short* __restrict__ WhT_f,
       const unsigned short* __restrict__ WhT_b,
       const float* __restrict__ bhn_f,
       const float* __restrict__ bhn_b,
       float* __restrict__ h_f, float* __restrict__ h_b,
       unsigned short* __restrict__ hbf_f,
       unsigned short* __restrict__ hbf_b,
       float* __restrict__ hp_f, float* __restrict__ hp_b,
       float* __restrict__ out, unsigned* bar) {
    const unsigned nb = SCAN_BLOCKS_PER_DIR;
    const int dir = (blockIdx.x >= (int)nb) ? 1 : 0;
    const int blk = blockIdx.x - dir * nb;

    const float* xp = dir ? xp_b : xp_f;
    const unsigned short* WhT = dir ? WhT_b : WhT_f;
    const float* bhn = dir ? bhn_b : bhn_f;
    float* h = dir ? h_b : h_f;
    unsigned short* hbf = dir ? hbf_b : hbf_f;
    float* hp = dir ? hp_b : hp_f;
    unsigned* cnt = bar + dir * 2;
    unsigned* gen = bar + dir * 2 + 1;

    const int lane = threadIdx.x & 31;
    const int wave = threadIdx.x >> 5;
    const int wid  = blk * 8 + wave;        // 0..383
    const int m0   = (wid & 3) * 16;        // batch-row tile
    const int n0   = (wid >> 2) * 16;       // gate-column tile
    const int tid  = blk * SCAN_THREADS + threadIdx.x;   // 0..12287 per dir
    const int nthr = nb * SCAN_THREADS;

    // zero the recurrent state
    for (int i = tid; i < BATCH * HDIM; i += nthr) { h[i] = 0.0f; hbf[i] = 0; }

    // hoist Wh: 16 B fragments, register-resident for all timesteps
    const unsigned short* Bbase = WhT + (size_t)n0 * HDIM;
    v16bf breg[16];
#pragma unroll
    for (int q = 0; q < 16; ++q)
        breg[q] = load_b32x16(Bbase + q * 32, HDIM, lane);

    grid_barrier(cnt, gen, nb);

    const unsigned short* Abase = hbf + (size_t)m0 * HDIM;

    for (int t = 0; t < TLEN; ++t) {
        const int te = dir ? (TLEN - 1 - t) : t;

        // ---- hp tile = h_bf16[m0:m0+16, :] @ Wh[:, n0:n0+16] ----
        v8f acc = {};
        v16bf areg[4];
#pragma unroll
        for (int p = 0; p < 4; ++p)
            areg[p] = load_a16x32(Abase + p * 32, HDIM, lane);
#pragma unroll
        for (int q = 0; q < 16; ++q) {
            v16bf cur = areg[q & 3];
            if (q + 4 < 16)        // refill this slot 4 WMMAs ahead
                areg[q & 3] = load_a16x32(Abase + (q + 4) * 32, HDIM, lane);
            acc = __builtin_amdgcn_wmma_f32_16x16x32_bf16(false, cur, false,
                                                          breg[q], (short)0,
                                                          acc, false, false);
        }
        {
            const int col = n0 + (lane & 15);
            const int rb  = m0 + ((lane >> 4) << 3);
#pragma unroll
            for (int v = 0; v < 8; ++v)
                hp[(size_t)(rb + v) * N3H + col] = acc[v];
        }
        grid_barrier(cnt, gen, nb);

        // ---- gates + state update ----
        for (int i = tid; i < BATCH * HDIM; i += nthr) {
            const int bb = i >> 9;          // batch
            const int jj = i & 511;         // hidden index
            const float* xrow = xp + ((size_t)bb * TLEN + te) * N3H;
            const float* hrow = hp + (size_t)bb * N3H;
            const float xr = xrow[jj], xz = xrow[HDIM + jj], xn = xrow[2 * HDIM + jj];
            const float hr = hrow[jj], hz = hrow[HDIM + jj], hn = hrow[2 * HDIM + jj];
            const float r = 1.0f / (1.0f + expf(-(xr + hr)));
            const float z = 1.0f / (1.0f + expf(-(xz + hz)));
            const float n = tanhf(xn + r * (hn + bhn[jj]));
            const float hnew = (1.0f - z) * n + z * h[i];
            h[i]   = hnew;
            hbf[i] = f32_to_bf16_rne(hnew);
            out[((size_t)bb * TLEN + te) * (2 * HDIM) + dir * HDIM + jj] = hnew;
        }
        grid_barrier(cnt, gen, nb);
    }
}

// ------------------------- host launcher -----------------------------------

extern "C" void kernel_launch(void* const* d_in, const int* in_sizes, int n_in,
                              void* d_out, int out_size, void* d_ws, size_t ws_size,
                              hipStream_t stream) {
    (void)in_sizes; (void)n_in; (void)out_size; (void)ws_size;
    const float* x     = (const float*)d_in[0];
    const float* Wi_f  = (const float*)d_in[1];
    const float* bi_f  = (const float*)d_in[2];
    const float* Wh_f  = (const float*)d_in[3];
    const float* bhn_f = (const float*)d_in[4];
    const float* Wi_b  = (const float*)d_in[5];
    const float* bi_b  = (const float*)d_in[6];
    const float* Wh_b  = (const float*)d_in[7];
    const float* bhn_b = (const float*)d_in[8];
    float* out = (float*)d_out;

    // workspace carve-up (256-byte aligned), ~444 MB total
    char* ws = (char*)d_ws;
    size_t off = 0;
    auto take = [&](size_t bytes) -> char* {
        char* p = ws + off;
        off = (off + bytes + 255) & ~(size_t)255;
        return p;
    };
    unsigned*       bar   = (unsigned*)take(64);
    unsigned short* xbf   = (unsigned short*)take((size_t)BT * DDIM * 2);
    unsigned short* WiT_f = (unsigned short*)take((size_t)N3H * DDIM * 2);
    unsigned short* WiT_b = (unsigned short*)take((size_t)N3H * DDIM * 2);
    unsigned short* WhT_f = (unsigned short*)take((size_t)N3H * HDIM * 2);
    unsigned short* WhT_b = (unsigned short*)take((size_t)N3H * HDIM * 2);
    float*          xp_f  = (float*)take((size_t)BT * N3H * 4);
    float*          xp_b  = (float*)take((size_t)BT * N3H * 4);
    float*          h_f   = (float*)take((size_t)BATCH * HDIM * 4);
    float*          h_b   = (float*)take((size_t)BATCH * HDIM * 4);
    unsigned short* hbf_f = (unsigned short*)take((size_t)BATCH * HDIM * 2);
    unsigned short* hbf_b = (unsigned short*)take((size_t)BATCH * HDIM * 2);
    float*          hp_f  = (float*)take((size_t)BATCH * N3H * 4);
    float*          hp_b  = (float*)take((size_t)BATCH * N3H * 4);

    k_init_bar<<<1, 32, 0, stream>>>(bar);
    k_cvt_bf16<<<4096, 256, 0, stream>>>(x, xbf, BT * DDIM);
    k_cvt_transpose<<<768, 256, 0, stream>>>(Wi_f, WiT_f, DDIM, N3H);
    k_cvt_transpose<<<768, 256, 0, stream>>>(Wi_b, WiT_b, DDIM, N3H);
    k_cvt_transpose<<<768, 256, 0, stream>>>(Wh_f, WhT_f, HDIM, N3H);
    k_cvt_transpose<<<768, 256, 0, stream>>>(Wh_b, WhT_b, HDIM, N3H);

    k_xproj<<<dim3(BT / (MT_PER_WAVE * 16) * NTILES_N / 8, 1, 2), 256, 0, stream>>>(
        xbf, WiT_f, WiT_b, bi_f, bi_b, xp_f, xp_b);

    k_scan<<<2 * SCAN_BLOCKS_PER_DIR, SCAN_THREADS, 0, stream>>>(
        xp_f, xp_b, WhT_f, WhT_b, bhn_f, bhn_b,
        h_f, h_b, hbf_f, hbf_b, hp_f, hp_b, out, bar);
}
